// PatchCrossAttention_28020366639426
// MI455X (gfx1250) — matrix operand order repaired and checked by reading, not compiled, not run on hardware
//
#include <hip/hip_runtime.h>

// ---------------- problem constants ----------------
#define NHEADS 8
#define HD     64
#define CDIM   512
#define NSEQ   1024
#define NBATCH 8

typedef __attribute__((ext_vector_type(16))) __bf16 v16bf;
typedef __attribute__((ext_vector_type(8)))  __bf16 v8bf;
typedef __attribute__((ext_vector_type(8)))  float  v8f;
typedef __attribute__((ext_vector_type(4)))  float  v4f;

static __device__ __forceinline__ v8f wmma_bf16(v16bf a, v16bf b, v8f c) {
  // D(16x16 f32) = A(16x32 bf16) * B(32x16 bf16) + C
  return __builtin_amdgcn_wmma_f32_16x16x32_bf16(false, a, false, b, (short)0, c,
                                                 false, false);
}

// A-fragment (16x32 bf16). Lane L: M = L&15 (caller passes row pointer), hg = L>>4.
// a[0..7] <-> K = k0 + 16*hg + i ; a[8..15] <-> K = k0 + 16*hg + 8 + i
static __device__ __forceinline__ v16bf load_a_bf16(const __bf16* row, int k0, int hg) {
  const __bf16* p = row + k0 + hg * 16;
  v8bf lo = *(const v8bf*)p;
  v8bf hi = *(const v8bf*)(p + 8);
  v16bf a;
#pragma unroll
  for (int i = 0; i < 8; ++i) { a[i] = lo[i]; a[8 + i] = hi[i]; }
  return a;
}

// ---------------- f32 -> bf16 conversion (one-time) ----------------
__global__ __launch_bounds__(256)
void cvt_bf16_kernel(const float* __restrict__ src, __bf16* __restrict__ dst, int n8) {
  const int i = blockIdx.x * 256 + threadIdx.x;
  if (i >= n8) return;
  const v4f* s = (const v4f*)src + (size_t)i * 2;
  v4f x0 = s[0];
  v4f x1 = s[1];
  v8bf d;
#pragma unroll
  for (int j = 0; j < 4; ++j) { d[j] = (__bf16)x0[j]; d[4 + j] = (__bf16)x1[j]; }
  *((v8bf*)dst + i) = d;
}

// ---------------- generic projection GEMM (bf16 in, pipelined) ----------------
// out[r, co] = (sum_k A[r,k] * W[co,k] + bias[co]) * scale
// mode 0: fp32 flat (R x 512) to outF
// mode 1: bf16 to outB at ((b*8+h)<<16) + n*sn + d*sd  (r=b*1024+n, co=h*64+d)
__global__ __launch_bounds__(128)
void gemm_wmma_kernel(const __bf16* __restrict__ A, const __bf16* __restrict__ W,
                      const float* __restrict__ bias, float scale, int mode,
                      float* __restrict__ outF, __bf16* __restrict__ outB,
                      int sn, int sd) {
  const int lane = threadIdx.x & 31;
  const int wv   = threadIdx.x >> 5;
  const int m0   = (blockIdx.x * 4 + wv) * 16;
  const int c0   = blockIdx.y * 64;
  const int nl   = lane & 15;
  const int hg   = lane >> 4;

  const __bf16* arow = A + (size_t)(m0 + nl) * CDIM;  // this lane's A row (M = nl)
  // B[k][n] = W[(col)*512 + k]; lane holds col n = nl, K = k0 + hg*16 + i (contig)
  const __bf16* wc[4];
#pragma unroll
  for (int t = 0; t < 4; ++t)
    wc[t] = W + (size_t)(c0 + t * 16 + nl) * CDIM + hg * 16;

  v8f acc[4] = {};

  // software pipeline: prefetch next K-step before issuing current WMMAs
  v16bf af = load_a_bf16(arow, 0, hg);
  v16bf b0 = *(const v16bf*)(wc[0]);
  v16bf b1 = *(const v16bf*)(wc[1]);
  v16bf b2 = *(const v16bf*)(wc[2]);
  v16bf b3 = *(const v16bf*)(wc[3]);

  for (int k0 = 0; k0 < CDIM - 32; k0 += 32) {
    const int kn = k0 + 32;
    v16bf afn = load_a_bf16(arow, kn, hg);
    v16bf n0 = *(const v16bf*)(wc[0] + kn);
    v16bf n1 = *(const v16bf*)(wc[1] + kn);
    v16bf n2 = *(const v16bf*)(wc[2] + kn);
    v16bf n3 = *(const v16bf*)(wc[3] + kn);
    acc[0] = wmma_bf16(af, b0, acc[0]);
    acc[1] = wmma_bf16(af, b1, acc[1]);
    acc[2] = wmma_bf16(af, b2, acc[2]);
    acc[3] = wmma_bf16(af, b3, acc[3]);
    af = afn; b0 = n0; b1 = n1; b2 = n2; b3 = n3;
  }
  acc[0] = wmma_bf16(af, b0, acc[0]);
  acc[1] = wmma_bf16(af, b1, acc[1]);
  acc[2] = wmma_bf16(af, b2, acc[2]);
  acc[3] = wmma_bf16(af, b3, acc[3]);

#pragma unroll
  for (int t = 0; t < 4; ++t) {
    const int col = c0 + t * 16 + nl;
    const float bv = bias[col];
#pragma unroll
    for (int r = 0; r < 8; ++r) {
      const int row = m0 + r + 8 * hg;            // D: M = r + 8*(lane>>4)
      const float v = (acc[t][r] + bv) * scale;
      if (mode == 0) {
        outF[(size_t)row * CDIM + col] = v;
      } else {
        const int bb = row >> 10, n = row & 1023;
        const int hh = col >> 6,  d = col & 63;
        outB[(((size_t)(bb * NHEADS + hh)) << 16) + (size_t)n * sn + (size_t)d * sd] =
            (__bf16)v;
      }
    }
  }
}

// ---------------- fused flash attention ----------------
// One wave handles 16 query rows of one (b,h). Online softmax; scores stay on-chip.
__global__ __launch_bounds__(128)
void attn_kernel(const __bf16* __restrict__ Qb, const __bf16* __restrict__ Kb,
                 const __bf16* __restrict__ Vt, const float* __restrict__ table,
                 const int* __restrict__ cur_pos, const int* __restrict__ prev_pos,
                 const int* __restrict__ mask, __bf16* __restrict__ X) {
  __shared__ __align__(32) float  sS[4][16][32];
  __shared__ __align__(32) __bf16 sP[4][16][32];
  __shared__ float sMax[4][16], sSum[4][16], sCorr[4][16];

  const int lane = threadIdx.x & 31;
  const int wv   = threadIdx.x >> 5;
  const int nl   = lane & 15;
  const int hg   = lane >> 4;
  const int bh   = blockIdx.y;
  const int b    = bh >> 3;
  const int h    = bh & 7;
  const int q0   = (blockIdx.x * 4 + wv) * 16;

  const __bf16* Qh = Qb + ((size_t)bh << 16);
  const __bf16* Kh = Kb + ((size_t)bh << 16);
  const __bf16* Vh = Vt + ((size_t)bh << 16);

  // Q A-fragments for K-dim 0..31 and 32..63 (Q already carries 1/sqrt(64))
  const __bf16* qrowp = Qh + (size_t)(q0 + nl) * HD;
  v16bf qa0 = load_a_bf16(qrowp, 0, hg);
  v16bf qa1 = load_a_bf16(qrowp, 32, hg);

  // hoist cur_pos for this lane's 8 D-fragment rows (M = r + 8*hg)
  int cpx[8], cpy[8];
#pragma unroll
  for (int r = 0; r < 8; ++r) {
    const int qrow = q0 + r + 8 * hg;
    cpx[r] = cur_pos[qrow * 2 + 0];
    cpy[r] = cur_pos[qrow * 2 + 1];
  }

  v8f o[4] = {};
  if (lane < 16) { sMax[wv][lane] = -1e30f; sSum[wv][lane] = 0.f; }
  __syncthreads();

  for (int kc = 0; kc < NSEQ; kc += 32) {
    // ---- issue ALL loads for this 32-key step up front ----
    const int kcol0 = kc + nl;        // chunk 0 key index for this lane
    const int kcol1 = kc + 16 + nl;   // chunk 1 key index
    const __bf16* kr0 = Kh + (size_t)kcol0 * HD;
    const __bf16* kr1 = Kh + (size_t)kcol1 * HD;
    v16bf kb00 = *(const v16bf*)(kr0 + hg * 16);
    v16bf kb01 = *(const v16bf*)(kr0 + 32 + hg * 16);
    v16bf kb10 = *(const v16bf*)(kr1 + hg * 16);
    v16bf kb11 = *(const v16bf*)(kr1 + 32 + hg * 16);
    v16bf vb[4];
#pragma unroll
    for (int t = 0; t < 4; ++t)   // V for phase 3: long time to cover latency
      vb[t] = *(const v16bf*)(Vh + (size_t)(t * 16 + nl) * NSEQ + kc + hg * 16);
    const int ppx0 = prev_pos[kcol0 * 2 + 0], ppy0 = prev_pos[kcol0 * 2 + 1];
    const int ppx1 = prev_pos[kcol1 * 2 + 0], ppy1 = prev_pos[kcol1 * 2 + 1];

    // ---- S = Q @ K^T (two 16-wide D tiles) ----
    v8f s0 = {}, s1 = {};
    s0 = wmma_bf16(qa0, kb00, s0);
    s0 = wmma_bf16(qa1, kb01, s0);
    s1 = wmma_bf16(qa0, kb10, s1);
    s1 = wmma_bf16(qa1, kb11, s1);

    // ---- bias + mask in-register, spill tiles to LDS for row softmax ----
#pragma unroll
    for (int r = 0; r < 8; ++r) {
      const int qrow = q0 + r + 8 * hg;
      const size_t mrow = ((size_t)b * NSEQ + qrow) * NSEQ;
      const int mv0 = mask[mrow + kcol0];
      const int mv1 = mask[mrow + kcol1];
      float sv0 = -1e30f, sv1 = -1e30f;
      if (mv0 != 0) {
        const int idx = (cpx[r] - ppx0 + 30) * 61 + (cpy[r] - ppy0 + 30);
        sv0 = s0[r] + table[idx * NHEADS + h];
      }
      if (mv1 != 0) {
        const int idx = (cpx[r] - ppx1 + 30) * 61 + (cpy[r] - ppy1 + 30);
        sv1 = s1[r] + table[idx * NHEADS + h];
      }
      sS[wv][r + 8 * hg][nl]      = sv0;
      sS[wv][r + 8 * hg][16 + nl] = sv1;
    }
    __syncthreads();

    // ---- online softmax: one lane per query row ----
    if (lane < 16) {
      const int m = lane;
      const float oldm = sMax[wv][m];
      float cm = oldm;
#pragma unroll
      for (int j = 0; j < 32; ++j) cm = fmaxf(cm, sS[wv][m][j]);
      const float corr = __expf(oldm - cm);
      float sum = sSum[wv][m] * corr;
#pragma unroll
      for (int j = 0; j < 32; ++j) {
        const float p = __expf(sS[wv][m][j] - cm);
        sP[wv][m][j] = (__bf16)p;
        sum += p;
      }
      sMax[wv][m] = cm; sSum[wv][m] = sum; sCorr[wv][m] = corr;
    }
    __syncthreads();

    // ---- O = diag(corr)*O + P @ V ----
    v16bf pa = load_a_bf16(&sP[wv][nl][0], 0, hg);         // P as A-fragment
    float cr[8];
#pragma unroll
    for (int r = 0; r < 8; ++r) cr[r] = sCorr[wv][r + 8 * hg];
#pragma unroll
    for (int t = 0; t < 4; ++t) {
#pragma unroll
      for (int r = 0; r < 8; ++r) o[t][r] *= cr[r];
      o[t] = wmma_bf16(pa, vb[t], o[t]);
    }
    __syncthreads();
  }

  // ---- normalize and write X (B, Nc, C) in bf16 for the final projection ----
  float inv[8];
#pragma unroll
  for (int r = 0; r < 8; ++r) inv[r] = 1.0f / sSum[wv][r + 8 * hg];
#pragma unroll
  for (int t = 0; t < 4; ++t) {
#pragma unroll
    for (int r = 0; r < 8; ++r) {
      const int qrow = q0 + r + 8 * hg;
      X[((size_t)b * NSEQ + qrow) * CDIM + h * HD + t * 16 + nl] =
          (__bf16)(o[t][r] * inv[r]);
    }
  }
}

// ---------------- host launch ----------------
extern "C" void kernel_launch(void* const* d_in, const int* in_sizes, int n_in,
                              void* d_out, int out_size, void* d_ws, size_t ws_size,
                              hipStream_t stream) {
  const float* cur     = (const float*)d_in[0];
  const float* prev    = (const float*)d_in[1];
  const float* q_w     = (const float*)d_in[2];
  const float* q_b     = (const float*)d_in[3];
  const float* kv_w    = (const float*)d_in[4];
  const float* kv_b    = (const float*)d_in[5];
  const float* proj_w  = (const float*)d_in[6];
  const float* proj_b  = (const float*)d_in[7];
  const float* table   = (const float*)d_in[8];
  const int*   cur_pos = (const int*)d_in[9];
  const int*   prev_pos= (const int*)d_in[10];
  const int*   maskp   = (const int*)d_in[11];

  char* ws = (char*)d_ws;
  const size_t MB = (size_t)1 << 20;
  __bf16* Qb    = (__bf16*)(ws);             // (B,H,Nc,64) bf16 : 8 MiB
  __bf16* Kb    = (__bf16*)(ws + 8  * MB);   // (B,H,Np,64) bf16 : 8 MiB
  __bf16* Vt    = (__bf16*)(ws + 16 * MB);   // (B,H,64,Np) bf16 : 8 MiB
  __bf16* Xb    = (__bf16*)(ws + 24 * MB);   // (B,Nc,C)    bf16 : 8 MiB
  __bf16* curb  = (__bf16*)(ws + 32 * MB);   // cur  bf16        : 8 MiB
  __bf16* prevb = (__bf16*)(ws + 40 * MB);   // prev bf16        : 8 MiB
  __bf16* wqb   = (__bf16*)(ws + 48 * MB);   // q_w  bf16        : 0.5 MiB
  __bf16* wkvb  = (__bf16*)(ws + 49 * MB);   // kv_w bf16        : 1 MiB
  __bf16* wprb  = (__bf16*)(ws + 51 * MB);   // proj_w bf16      : 0.5 MiB
  float*  out   = (float*)d_out;

  // one-time f32 -> bf16 conversions (all L2-resident afterwards)
  const int nAct = NBATCH * NSEQ * CDIM;           // 4M
  const int nW1  = CDIM * CDIM;                    // 256K
  const int nW2  = 2 * CDIM * CDIM;                // 512K
  cvt_bf16_kernel<<<(nAct / 8 + 255) / 256, 256, 0, stream>>>(cur,    curb,  nAct / 8);
  cvt_bf16_kernel<<<(nAct / 8 + 255) / 256, 256, 0, stream>>>(prev,   prevb, nAct / 8);
  cvt_bf16_kernel<<<(nW1  / 8 + 255) / 256, 256, 0, stream>>>(q_w,    wqb,   nW1 / 8);
  cvt_bf16_kernel<<<(nW2  / 8 + 255) / 256, 256, 0, stream>>>(kv_w,   wkvb,  nW2 / 8);
  cvt_bf16_kernel<<<(nW1  / 8 + 255) / 256, 256, 0, stream>>>(proj_w, wprb,  nW1 / 8);

  dim3 gemm_grid(128, 8), blk(128);
  // Q = cur @ q_w^T + q_b, scaled by hd^-0.5 -> (B,H,Nc,64)
  gemm_wmma_kernel<<<gemm_grid, blk, 0, stream>>>(curb, wqb, q_b, 0.125f, 1,
                                                  nullptr, Qb, 64, 1);
  // K = prev @ kv_w[0:512]^T + kv_b[0:512] -> (B,H,Np,64)
  gemm_wmma_kernel<<<gemm_grid, blk, 0, stream>>>(prevb, wkvb, kv_b, 1.0f, 1,
                                                  nullptr, Kb, 64, 1);
  // V = prev @ kv_w[512:1024]^T + kv_b[512:1024] -> transposed (B,H,64,Np)
  gemm_wmma_kernel<<<gemm_grid, blk, 0, stream>>>(prevb, wkvb + (size_t)512 * 512,
                                                  kv_b + 512, 1.0f, 1,
                                                  nullptr, Vt, 1, 1024);
  // fused attention (writes bf16 X)
  attn_kernel<<<dim3(16, 64), blk, 0, stream>>>(Qb, Kb, Vt, table, cur_pos,
                                                prev_pos, maskp, Xb);
  // out = X @ proj_w^T + proj_b (fp32 output)
  gemm_wmma_kernel<<<gemm_grid, blk, 0, stream>>>(Xb, wprb, proj_b, 1.0f, 0,
                                                  out, nullptr, 0, 0);
}
